// DotProductAttention_46823733461592
// MI455X (gfx1250) — matrix operand order repaired and checked by reading, not compiled, not run on hardware
//
#include <hip/hip_runtime.h>
#include <math.h>

// ---------- types ----------
typedef __attribute__((ext_vector_type(16))) __bf16          v16bf;
typedef __attribute__((ext_vector_type(8)))  float           v8f;
typedef __attribute__((ext_vector_type(8)))  unsigned short  ushort8;
typedef __attribute__((ext_vector_type(16))) unsigned short  ushort16;
typedef __attribute__((ext_vector_type(4)))  unsigned int    uint4v;
typedef __attribute__((ext_vector_type(8)))  int             int8v;
typedef __attribute__((ext_vector_type(4)))  int             int4v;
typedef unsigned short     u16;
typedef unsigned int       u32;
typedef unsigned long long u64;

#if __has_builtin(__builtin_amdgcn_tensor_load_to_lds)
#define HAVE_TDM 1
#else
#define HAVE_TDM 0
#endif

// ---------- problem dims ----------
#define NBATCH 32
#define CC     256
#define TTX    1024
#define TTY    1024

// ---------- GEMM tiling ----------
#define BM 128
#define BN 128
#define BK 32
#define LDSS 40   // 32 + 8 halfword pad -> 80B row stride (16B aligned)

__device__ __forceinline__ u16 f32_to_bf16(float f) {
    return __builtin_bit_cast(u16, (__bf16)f);   // v_cvt_pk_bf16_f32 pairs
}

__device__ __forceinline__ float fast_tanh(float x) {
#if __has_builtin(__builtin_amdgcn_tanhf)
    return __builtin_amdgcn_tanhf(x);            // native v_tanh_f32
#else
    return tanhf(x);
#endif
}

union Frag16 { ushort16 v; ushort8 h[2]; };

#if HAVE_TDM
// One-shot TDM: copy a (tile_d1 rows x tile_d0 elems) bf16 tile, row stride
// `stride` (elems), from global into LDS rows padded 64B data + 16B pad
// (pad_interval=3 -> every 16 DWORDs, pad_amount=3 -> 4 DWORDs) == LDSS layout.
__device__ __forceinline__ void tdm_load_2d(u32 lds_addr, const u16* gsrc,
                                            u32 tile_d0, u32 tile_d1, u64 stride)
{
    const u64 ga = (u64)gsrc;
    uint4v g0;
    g0[0] = 1u;                                           // count=1 (valid), user mode
    g0[1] = lds_addr;                                     // lds_addr [63:32]
    g0[2] = (u32)ga;                                      // global_addr low
    g0[3] = ((u32)(ga >> 32) & 0x01FFFFFFu) | (2u << 30); // global_addr[56:32] | type=2
    int8v g1;
    g1[0] = (int)((1u << 16)      // data_size = 2 bytes
                | (1u << 20)      // pad_enable
                | (3u << 22)      // pad_interval: every 16 DWORDs (64B)
                | (3u << 25));    // pad_amount: 4 DWORDs (16B)
    g1[1] = (int)((tile_d0 & 0xFFFFu) << 16);                         // tensor_dim0 lo16
    g1[2] = (int)(((tile_d0 >> 16) & 0xFFFFu) | ((tile_d1 & 0xFFFFu) << 16)); // d0 hi | d1 lo
    g1[3] = (int)(((tile_d1 >> 16) & 0xFFFFu) | ((tile_d0 & 0xFFFFu) << 16)); // d1 hi | tile_dim0
    g1[4] = (int)(tile_d1 & 0xFFFFu);                                 // tile_dim1 (tile_dim2=0)
    g1[5] = (int)(u32)stride;                                         // dim0_stride lo32
    g1[6] = (int)((u32)(stride >> 32) & 0xFFFFu);                     // dim0_stride hi16
    g1[7] = 0;
    const int4v z4 = {0, 0, 0, 0};
    const int8v z8 = {0, 0, 0, 0, 0, 0, 0, 0};
    __builtin_amdgcn_tensor_load_to_lds(g0, g1, z4, z4, z8, 0);
}
#endif

// D = epilogue( A * B ), bf16 WMMA with f32 accumulate.
//   A: (M x K), (m,k) at A + m*sAm + k [k contig]; AF32 ? f32 converted : bf16 via TDM/async
//   B: BT ? bf16 transposed (N x K), (n,k) at B + n*sB + k [k contig, TDM/async]
//        : f32 (K x N), (k,n) at B + k*sB + n [n contig, transposed while staging]
// EPI: 0 = f32 store   1 = f32 store * scale
//      2 = bf16 tanh(x+bias[m])   3 = bf16 tanh(x+bias[m]) transposed (Out[n*ldO+m])
template<bool AF32, bool BT, int EPI>
__global__ void __launch_bounds__(256)
gemm_wmma(const void* __restrict__ Av, const void* __restrict__ Bv,
          void* __restrict__ Ov, const float* __restrict__ bias,
          int K, long sAm, long sB, long ldO,
          long batA, long batB, long batO, float scale)
{
    __shared__ __align__(16) u16 As[2][BM][LDSS];
    __shared__ __align__(16) u16 Bs[2][BN][LDSS];

    const int tid  = threadIdx.x;
    const int lane = tid & 31;
    const int wid  = tid >> 5;
    const int wm   = wid & 3;       // 4 waves x 32 rows = 128 M
    const int wn   = wid >> 2;      // 2 waves x 64 cols = 128 N
    const int lr   = lane & 15;
    const int half = lane >> 4;

    const int bm = blockIdx.y * BM;
    const int bn = blockIdx.x * BN;
    const int bz = blockIdx.z;

    const float* Af = (const float*)Av + (long)bz * batA;
    const u16*   Ah = (const u16*)  Av + (long)bz * batA;
    const float* Bf = (const float*)Bv + (long)bz * batB;
    const u16*   Bh = (const u16*)  Bv + (long)bz * batB;

    // f32-A loader mapping: row am = tid/2, 16 contiguous k at (tid&1)*16
    const int am = tid >> 1;
    const int ak = (tid & 1) * 16;
    // f32-B loader mapping: k-row = tid/8, 16 contiguous n at (tid&7)*16
    const int bkr = tid >> 3;
    const int bns = (tid & 7) * 16;

    auto load_stage = [&](int buf, int k0) {
        // ---------------- A tile -> LDS [m][k] ----------------
        if (AF32) {
            const float* p = Af + (long)(bm + am) * sAm + (k0 + ak);
            u16* dst = &As[buf][am][ak];
            #pragma unroll
            for (int v = 0; v < 4; ++v) {
                float4 f = *(const float4*)(p + v * 4);
                dst[v*4+0] = f32_to_bf16(f.x);
                dst[v*4+1] = f32_to_bf16(f.y);
                dst[v*4+2] = f32_to_bf16(f.z);
                dst[v*4+3] = f32_to_bf16(f.w);
            }
        } else {
#if HAVE_TDM
            if (wid == 0)
                tdm_load_2d((u32)(u64)&As[buf][0][0],
                            Ah + (long)bm * sAm + k0, BK, BM, (u64)sAm);
#else
            #pragma unroll
            for (int i = 0; i < 2; ++i) {
                const int c   = tid + i * 256;
                const int row = c >> 2, seg = c & 3;
                u32 dst = (u32)(u64)&As[buf][row][seg * 8];
                u64 src = (u64)(Ah + (long)(bm + row) * sAm + k0 + seg * 8);
                asm volatile("global_load_async_to_lds_b128 %0, %1, off"
                             :: "v"(dst), "v"(src) : "memory");
            }
#endif
        }
        // ---------------- B tile -> LDS [n][k] ----------------
        if (BT) {
#if HAVE_TDM
            if (wid == 0)
                tdm_load_2d((u32)(u64)&Bs[buf][0][0],
                            Bh + (long)bn * sB + k0, BK, BN, (u64)sB);
#else
            #pragma unroll
            for (int i = 0; i < 2; ++i) {
                const int c   = tid + i * 256;
                const int row = c >> 2, seg = c & 3;
                u32 dst = (u32)(u64)&Bs[buf][row][seg * 8];
                u64 src = (u64)(Bh + (long)(bn + row) * sB + k0 + seg * 8);
                asm volatile("global_load_async_to_lds_b128 %0, %1, off"
                             :: "v"(dst), "v"(src) : "memory");
            }
#endif
        } else if (!BT) {
            const float* p = Bf + (long)(k0 + bkr) * sB + (bn + bns);
            #pragma unroll
            for (int v = 0; v < 4; ++v) {
                float4 f = *(const float4*)(p + v * 4);
                Bs[buf][bns + v*4 + 0][bkr] = f32_to_bf16(f.x);
                Bs[buf][bns + v*4 + 1][bkr] = f32_to_bf16(f.y);
                Bs[buf][bns + v*4 + 2][bkr] = f32_to_bf16(f.z);
                Bs[buf][bns + v*4 + 3][bkr] = f32_to_bf16(f.w);
            }
        }
    };

    v8f acc[2][4] = {};

    load_stage(0, 0);
    const int nk = K / BK;
    for (int kt = 0; kt < nk; ++kt) {
        if (!AF32 || BT) {    // drain outstanding tile DMAs before the barrier
#if HAVE_TDM
            __builtin_amdgcn_s_wait_tensorcnt(0);
#else
            asm volatile("s_wait_asynccnt 0x0" ::: "memory");
#endif
        }
        __syncthreads();
        const int cur = kt & 1;
        if (kt + 1 < nk) load_stage(cur ^ 1, (kt + 1) * BK);

        // ---- fragments from LDS (16B ds_load_b128 each) ----
        Frag16 afr[2];
        #pragma unroll
        for (int i = 0; i < 2; ++i) {
            const int row = wm * 32 + i * 16 + lr;
            // 16-bit A layout: lane<16 -> K=0..7 & 16..23 ; lane>=16 -> K=8..15 & 24..31
            afr[i].h[0] = *(const ushort8*)&As[cur][row][half * 8];
            afr[i].h[1] = *(const ushort8*)&As[cur][row][16 + half * 8];
        }
        Frag16 bfr[4];
        #pragma unroll
        for (int j = 0; j < 4; ++j) {
            const int col = wn * 64 + j * 16 + lr;
            // B layout (matches A): lane<16 -> K=0..7 & 16..23 ; lane>=16 -> K=8..15 & 24..31
            bfr[j].h[0] = *(const ushort8*)&Bs[cur][col][half * 8];
            bfr[j].h[1] = *(const ushort8*)&Bs[cur][col][16 + half * 8];
        }
        #pragma unroll
        for (int i = 0; i < 2; ++i)
            #pragma unroll
            for (int j = 0; j < 4; ++j)
                acc[i][j] = __builtin_amdgcn_wmma_f32_16x16x32_bf16(
                    false, __builtin_bit_cast(v16bf, afr[i].v),
                    false, __builtin_bit_cast(v16bf, bfr[j].v),
                    (short)0, acc[i][j], false, false);
    }

    // ---- epilogue ----
    float* OutF = (float*)Ov + (long)bz * batO;
    u16*   OutH = (u16*)  Ov + (long)bz * batO;
    #pragma unroll
    for (int i = 0; i < 2; ++i) {
        #pragma unroll
        for (int j = 0; j < 4; ++j) {
            const int tm = bm + wm * 32 + i * 16 + half * 8;
            const int tn = bn + wn * 64 + j * 16 + lr;
            #pragma unroll
            for (int r = 0; r < 8; ++r) {
                const int m = tm + r;
                const float x = acc[i][j][r];
                if (EPI == 0) {
                    OutF[(long)m * ldO + tn] = x;
                } else if (EPI == 1) {
                    OutF[(long)m * ldO + tn] = x * scale;
                } else {
                    const float t = fast_tanh(x + bias[m]);
                    if (EPI == 2) OutH[(long)m * ldO + tn] = f32_to_bf16(t);
                    else          OutH[(long)tn * ldO + m] = f32_to_bf16(t);
                }
            }
        }
    }
}

// In-place column softmax over s (axis 0) of (TTX x TTY) f32 per batch.
// WT=true additionally writes bf16 A^T (TTY x TTX) for the TDM stage-4 GEMM.
template<bool WT>
__global__ void __launch_bounds__(256)
softmax_cols(float* __restrict__ P, u16* __restrict__ At)
{
    const int t = blockIdx.x * 256 + threadIdx.x;
    float* p = P + (long)blockIdx.z * (long)TTX * TTY + t;
    float m = -1e30f, l = 0.0f;
    for (int s = 0; s < TTX; ++s) {
        const float x = p[(long)s * TTY];
        const float nm = fmaxf(m, x);
        l = l * __expf(m - nm) + __expf(x - nm);
        m = nm;
    }
    const float inv = 1.0f / l;
    if (WT) {
        u16* arow = At + (long)blockIdx.z * (long)TTX * TTY + (long)t * TTX;
        for (int s0 = 0; s0 < TTX; s0 += 8) {
            ushort8 w;
            #pragma unroll
            for (int j = 0; j < 8; ++j) {
                const long idx = (long)(s0 + j) * TTY;
                const float e = __expf(p[idx] - m) * inv;
                p[idx] = e;
                w[j] = f32_to_bf16(e);
            }
            *(ushort8*)(arow + s0) = w;
        }
    } else {
        for (int s = 0; s < TTX; ++s) {
            const long idx = (long)s * TTY;
            p[idx] = __expf(p[idx] - m) * inv;
        }
    }
}

extern "C" void kernel_launch(void* const* d_in, const int* in_sizes, int n_in,
                              void* d_out, int out_size, void* d_ws, size_t ws_size,
                              hipStream_t stream)
{
    const float* Q  = (const float*)d_in[0];
    const float* K  = (const float*)d_in[1];
    const float* V  = (const float*)d_in[2];
    const float* Wq = (const float*)d_in[3];
    const float* bq = (const float*)d_in[4];
    const float* Wk = (const float*)d_in[5];
    const float* bk = (const float*)d_in[6];
    const float* Wv = (const float*)d_in[7];
    const float* bv = (const float*)d_in[8];

    float* outR = (float*)d_out;                       // (N, C, TY) f32
    float* outA = outR + (size_t)NBATCH * CC * TTY;    // (N, TX, TY) f32

    const size_t projElems = (size_t)NBATCH * CC * TTY;      // 8M per tensor
    const size_t atElems   = (size_t)NBATCH * TTX * TTY;     // 32M
    u16* qpt = (u16*)d_ws;              // (N, TY, C)  Qp transposed
    u16* kpt = qpt + projElems;         // (N, TX, C)  Kp transposed
    u16* vp  = kpt + projElems;         // (N, C, TX)  Vp
    u16* At  = vp + projElems;          // (N, TY, TX) bf16 softmax(A)^T (optional)
    const bool useAt = ws_size >= (3 * projElems + atElems) * sizeof(u16);

    dim3 blk(256);

    // Stage 1: projections (M=C, N=T, K=C), A=W f32, B=X f32
    dim3 gP(TTY / BN, CC / BM, NBATCH);
    gemm_wmma<true, false, 3><<<gP, blk, 0, stream>>>(     // Qp stored transposed (TY x C)
        Wq, Q, qpt, bq, CC,
        /*sAm*/CC, /*sB*/TTY, /*ldO*/CC,
        0, (long)CC * TTY, (long)TTY * CC, 1.0f);
    gemm_wmma<true, false, 3><<<gP, blk, 0, stream>>>(     // Kp stored transposed (TX x C)
        Wk, K, kpt, bk, CC,
        CC, TTX, CC,
        0, (long)CC * TTX, (long)TTX * CC, 1.0f);
    gemm_wmma<true, false, 2><<<gP, blk, 0, stream>>>(     // Vp natural (C x TX)
        Wv, V, vp, bv, CC,
        CC, TTX, TTX,
        0, (long)CC * TTX, (long)CC * TTX, 1.0f);

    // Stage 2: scores = Kp^T Qp / sqrt(C) -> outA   (M=TX, N=TY, K=C), both bf16 TDM
    dim3 gS(TTY / BN, TTX / BM, NBATCH);
    gemm_wmma<false, true, 1><<<gS, blk, 0, stream>>>(
        kpt, qpt, outA, nullptr, CC,
        /*sAm*/CC, /*sB*/CC, /*ldO*/TTY,
        (long)TTX * CC, (long)TTY * CC, (long)TTX * TTY, 0.0625f);

    // Stage 3: softmax over s (in place), optionally emit bf16 A^T
    dim3 gSm(TTY / 256, 1, NBATCH);
    if (useAt) softmax_cols<true ><<<gSm, blk, 0, stream>>>(outA, At);
    else       softmax_cols<false><<<gSm, blk, 0, stream>>>(outA, nullptr);

    // Stage 4: R = Vp A -> outR   (M=C, N=TY, K=TX)
    dim3 gR(TTY / BN, CC / BM, NBATCH);
    if (useAt)
        gemm_wmma<false, true, 0><<<gR, blk, 0, stream>>>(   // both operands bf16 TDM
            vp, At, outR, nullptr, TTX,
            /*sAm*/TTX, /*sB*/TTX, /*ldO*/TTY,
            (long)CC * TTX, (long)TTX * TTY, (long)CC * TTY, 1.0f);
    else
        gemm_wmma<false, false, 0><<<gR, blk, 0, stream>>>(  // B = f32 A, converted on the fly
            vp, outA, outR, nullptr, TTX,
            TTX, /*sB (k-row stride)*/TTY, TTY,
            (long)CC * TTX, (long)TTX * TTY, (long)CC * TTY, 1.0f);
}
